// HybridReIDModel_67972152426661
// MI455X (gfx1250) — matrix operand order repaired and checked by reading, not compiled, not run on hardware
//
#include <hip/hip_runtime.h>
#include <hip/hip_bf16.h>

// ---------------- types for CDNA5 WMMA / TDM ----------------
typedef __bf16 bf16raw;
typedef bf16raw v16bf __attribute__((ext_vector_type(16)));
typedef float   v8f   __attribute__((ext_vector_type(8)));
typedef unsigned int u32x4 __attribute__((ext_vector_type(4)));
typedef int          i32x4 __attribute__((ext_vector_type(4)));
typedef int          i32x8 __attribute__((ext_vector_type(8)));

#define HAVE_TDM __has_builtin(__builtin_amdgcn_tensor_load_to_lds)

// ---------------- bf16 helpers (raw ushort storage) ----------------
__device__ __forceinline__ unsigned short f2bf(float f) {
  union { float f; unsigned u; } v; v.f = f;
  unsigned r = v.u + 0x7FFFu + ((v.u >> 16) & 1u);   // round-to-nearest-even
  return (unsigned short)(r >> 16);
}
__device__ __forceinline__ float bf2f(unsigned short h) {
  union { unsigned u; float f; } v; v.u = ((unsigned)h) << 16;
  return v.f;
}

// ---------------- f32 -> bf16 packing ----------------
__global__ void k_f32_to_bf16(const float* __restrict__ src,
                              unsigned short* __restrict__ dst, int n) {
  int i = blockIdx.x * 256 + threadIdx.x;
  if (i < n) dst[i] = f2bf(src[i]);
}

#if HAVE_TDM
// TDM: async DMA of a 2-D tile (tile_rows x tile_cols, 2-byte elems, row stride
// in elements) from global memory into LDS. D# per CDNA5 ISA.
__device__ __forceinline__ void tdm_load_2d_bf16(unsigned lds_addr, const void* gptr,
                                                 int tile_cols, int tile_rows,
                                                 long long row_stride_elems) {
  const unsigned long long ga = (unsigned long long)gptr;
  u32x4 g0 = {};
  g0[0] = 1u;                                             // count=1, user mode
  g0[1] = lds_addr;                                       // LDS byte address
  g0[2] = (unsigned)ga;                                   // global_addr[31:0]
  g0[3] = (unsigned)((ga >> 32) & 0x01ffffffu) | (2u << 30);  // addr[56:32]|type=2
  i32x8 g1 = {};
  g1[0] = 0x00010000;                                     // data_size=1 (2 bytes)
  g1[1] = (tile_cols & 0xffff) << 16;                     // tensor_dim0[15:0]
  g1[2] = (int)(((unsigned)tile_cols >> 16) |             // tensor_dim0[31:16]
                ((unsigned)(tile_rows & 0xffff) << 16));  // tensor_dim1[15:0]
  g1[3] = (int)(((unsigned)tile_rows >> 16) |             // tensor_dim1[31:16]
                ((unsigned)(tile_cols & 0xffff) << 16));  // tile_dim0
  g1[4] = tile_rows & 0xffff;                             // tile_dim1 (tile_dim2=0)
  g1[5] = (int)(row_stride_elems & 0xffffffffll);         // dim0_stride[31:0]
  g1[6] = (int)((row_stride_elems >> 32) & 0xffffll);     // dim0_stride[47:32]
  g1[7] = 0;
  i32x4 z4 = {};
#if __has_include(<hip/amd_detail/amd_gfx1250_TDM.h>)
  i32x8 z8 = {};
  __builtin_amdgcn_tensor_load_to_lds(g0, g1, z4, z4, z8, 0);
#else
  __builtin_amdgcn_tensor_load_to_lds(g0, g1, z4, z4, 0);
#endif
}
#endif

// ---------------- implicit-GEMM conv + BN (+residual)(+ReLU), bf16 WMMA ----------------
// GEMM view: M = B*Ho*Wo, N = Cout, K = Cin*KH*KW (OIHW flatten for weights).
// Block tile: 128(M) x 64(N); 8 waves; wave w owns N-tile (w&3), M-tiles [(w>>2)*64, +64).
// B tile staged by the Tensor Data Mover when K%32==0; A tile is on-the-fly im2col.
// All K-step filter decode uses 16-bit magic reciprocals (exact: kg < 2^16).
__global__ void __launch_bounds__(256)
k_conv_bn_wmma(const unsigned short* __restrict__ in,   // bf16 [B][Ci][H][W]
               const unsigned short* __restrict__ wgt,  // bf16 [Co][K]
               const float* __restrict__ bng, const float* __restrict__ bnb,
               const float* __restrict__ bnm, const float* __restrict__ bnv,
               const unsigned short* __restrict__ res,  // bf16 residual or null
               unsigned short* __restrict__ out,        // bf16 [B][Co][Ho][Wo]
               int Ci, int H, int W, int Co, int Ho, int Wo,
               int KH, int KW, int stride, int pad, int relu,
               int K, int Mtot, int use_tdm,
               unsigned mgKHW, unsigned mgKW) {         // ceil(2^16/KHW), ceil(2^16/KW)
  __shared__ __align__(16) unsigned short Asm[2][128][32];  // im2col tile [m][k]
  __shared__ __align__(16) unsigned short Bsm[2][64][32];   // weights [n][k]

  const int tid   = threadIdx.x;
  const int lane  = tid & 31;
  const int wave  = tid >> 5;
  const int m0    = blockIdx.y * 128;
  const int n0    = blockIdx.x * 64;
  const int HoWo  = Ho * Wo;
  const int KHW   = KH * KW;
  const int HW    = H * W;
  const int KT    = (K + 31) >> 5;
  const int mbase = (wave >> 2) * 64;
  const int ntile = wave & 3;

  // ---- hoisted per-thread A-row decode (kt-invariant) ----
  const int kl    = tid & 31;   // A k-lane: constant across the 16 row slots
  const int mrow0 = tid >> 5;
  int aoff[16];                 // b * Ci*H*W
  int sp[16];                   // (oh*stride)<<16 | (ow*stride)
  unsigned mmask = 0;
  for (int it = 0; it < 16; ++it) {
    const int m = m0 + mrow0 + (it << 3);
    const int b = m / HoWo, r2 = m - b * HoWo;
    const int oh = r2 / Wo,  ow = r2 - oh * Wo;
    if (m < Mtot) mmask |= (1u << it);
    aoff[it] = b * (Ci * HW);
    sp[it]   = ((oh * stride) << 16) | (ow * stride);
  }

  auto load_tiles = [&](int kt, int buf) {
    const int k0 = kt << 5;
#if HAVE_TDM
    if (use_tdm) {
      if (wave == 0)   // TDM ignores EXEC; issue once, uniform operands; overlaps gather
        tdm_load_2d_bf16((unsigned)(size_t)&Bsm[buf][0][0],
                         wgt + (size_t)n0 * K + k0, 32, 64, (long long)K);
    } else
#endif
    {
      for (int e = tid; e < 64 * 32; e += 256) {
        const int nl = e >> 5, kk = e & 31;
        const int n = n0 + nl, kg2 = k0 + kk;
        unsigned short v = 0;
        if (n < Co && kg2 < K) {
          v = wgt[(size_t)n * K + kg2];
          __builtin_prefetch(&wgt[(size_t)n * K + kg2 + 32], 0, 1);
        }
        Bsm[buf][nl][kk] = v;
      }
    }
    // branch-free filter-offset decode via 16-bit magic reciprocals
    const int kg = k0 + kl;
    const int ci = (int)(((unsigned)kg * mgKHW) >> 16);
    const int rs = kg - ci * KHW;
    const int dr = (int)(((unsigned)rs * mgKW) >> 16);
    const int dc = rs - dr * KW;
    const bool kval  = kg < K;
    const int  cioff = ci * HW;
    for (int it = 0; it < 16; ++it) {
      unsigned short v = 0;
      const int ih = (sp[it] >> 16) - pad + dr;
      const int iw = (sp[it] & 0xffff) - pad + dc;
      if (kval && ((mmask >> it) & 1) &&
          (unsigned)ih < (unsigned)H && (unsigned)iw < (unsigned)W)
        v = in[(size_t)(aoff[it] + cioff + ih * W + iw)];
      Asm[buf][mrow0 + (it << 3)][kl] = v;
    }
  };

  v8f acc[4];
  for (int i = 0; i < 4; ++i)
    for (int e = 0; e < 8; ++e) acc[i][e] = 0.f;

  load_tiles(0, 0);
  for (int kt = 0; kt < KT; ++kt) {
#if HAVE_TDM
    if (use_tdm && wave == 0) __builtin_amdgcn_s_wait_tensorcnt(0);
#endif
    __syncthreads();                         // tile kt visible; prev compute done
    const int cb = kt & 1;
    if (kt + 1 < KT) load_tiles(kt + 1, cb ^ 1);

    // Stage ALL fragments first, then burst the 4 WMMAs back-to-back so a single
    // dscnt wait covers the LDS reads and the XDL pipe runs without stalls.
    union { uint4 u[2]; v16bf v; } bu, au[4];
    // B fragment: elem i -> K = (lane/16)*16 + i, N = ntile*16 + lane%16
    const unsigned short* bp = &Bsm[cb][ntile * 16 + (lane & 15)][(lane >> 4) << 4];
    bu.u[0] = ((const uint4*)bp)[0];
    bu.u[1] = ((const uint4*)bp)[1];
#pragma unroll
    for (int mi = 0; mi < 4; ++mi) {
      // A fragment: elem i -> K = (lane/16)*16 + i, M = mtile*16 + lane%16
      const unsigned short* ap =
          &Asm[cb][mbase + mi * 16 + (lane & 15)][(lane >> 4) << 4];
      au[mi].u[0] = *(const uint4*)ap;
      au[mi].u[1] = *(const uint4*)(ap + 8);
    }
#pragma unroll
    for (int mi = 0; mi < 4; ++mi)
      acc[mi] = __builtin_amdgcn_wmma_f32_16x16x32_bf16(
          false, au[mi].v, false, bu.v, (short)0, acc[mi], false, false);
  }

  // Epilogue: BN is per-N and N is lane-invariant across the C fragment column.
  const int n = n0 + ntile * 16 + (lane & 15);
  float scale = 0.f, bias = 0.f;
  if (n < Co) {
    const float inv = rsqrtf(bnv[n] + 1e-5f);
    scale = bng[n] * inv;
    bias  = bnb[n] - bnm[n] * scale;
  }
  const bool nok = n < Co;
  const size_t nrow = (size_t)n * HoWo;
  const int hi8 = (lane & 16) ? 8 : 0;
#pragma unroll
  for (int mi = 0; mi < 4; ++mi) {
    // 8-row strip is 8-aligned and HoWo is a multiple of 32 -> never crosses a
    // batch boundary: decode (b, spatial) once per strip.
    const int mb = m0 + mbase + mi * 16 + hi8;
    if (mb < Mtot && nok) {
      const int b = mb / HoWo, r2b = mb - b * HoWo;
      const size_t base = (size_t)b * Co * HoWo + nrow + r2b;
#pragma unroll
      for (int r = 0; r < 8; ++r) {
        float y = acc[mi][r] * scale + bias;
        const size_t oi = base + r;
        if (res)  y += bf2f(res[oi]);
        if (relu) y = fmaxf(y, 0.f);
        out[oi] = f2bf(y);
      }
    }
  }
}

// ---------------- 3x3/s2/p1 maxpool: [32][64][128][64] -> [32][64][64][32] ----------------
__global__ void k_maxpool(const unsigned short* __restrict__ in,
                          unsigned short* __restrict__ out) {
  int i = blockIdx.x * 256 + threadIdx.x;
  const int total = 32 * 64 * 64 * 32;
  if (i >= total) return;
  int ow = i & 31, t = i >> 5;
  int oh = t & 63; t >>= 6;
  int c  = t & 63; t >>= 6;
  int b  = t;
  float m = -3.4e38f;
  for (int r = 0; r < 3; ++r)
    for (int s = 0; s < 3; ++s) {
      int ih = oh * 2 - 1 + r, iw = ow * 2 - 1 + s;
      if ((unsigned)ih < 128u && (unsigned)iw < 64u)
        m = fmaxf(m, bf2f(in[(((size_t)b * 64 + c) * 128 + ih) * 64 + iw]));
    }
  out[i] = f2bf(m);
}

// ---------------- global avgpool: bf16 [32][2048][8][4] -> f32 [32][2048] ----------------
__global__ void k_avgpool(const unsigned short* __restrict__ in,
                          float* __restrict__ feats) {
  int i = blockIdx.x * 256 + threadIdx.x;
  if (i >= 32 * 2048) return;
  const unsigned short* p = in + (size_t)i * 32;
  float s = 0.f;
  for (int j = 0; j < 32; ++j) s += bf2f(p[j]);
  feats[i] = s * (1.f / 32.f);
}

// ---------------- 8-qubit circuit sim: one wave32 per (circuit, batch) ----------------
// AmplitudeEmbedding(normalize), 8 layers of {RX(q); CNOT ring}, then <Z_q>.
// Wire 0 is the MSB of the basis index (bit position 7-q).
__global__ void __launch_bounds__(32)
k_qsim(const float* __restrict__ feats,   // [32][2048]
       const float* __restrict__ qw,      // [8][8][8]
       float* __restrict__ qout) {        // [32][64]
  __shared__ float sre[256], sim_[256];
  const int ql = blockIdx.x, b = blockIdx.y, lane = threadIdx.x;
  const float* xp = feats + (size_t)b * 2048 + (size_t)ql * 256;

  float ss = 0.f;
  for (int j = 0; j < 8; ++j) { float v = xp[lane * 8 + j]; ss += v * v; }
  for (int o = 16; o; o >>= 1) ss += __shfl_xor(ss, o, 32);
  const float rn = rsqrtf(ss + 1e-30f);
  for (int j = 0; j < 8; ++j) {
    sre[lane * 8 + j] = xp[lane * 8 + j] * rn;
    sim_[lane * 8 + j] = 0.f;
  }
  __syncthreads();

  const float* wq = qw + (size_t)ql * 64;
  for (int l = 0; l < 8; ++l) {
    for (int q = 0; q < 8; ++q) {               // RX rotations
      const float half = 0.5f * wq[l * 8 + q];
      const float c = cosf(half), s = sinf(half);
      const int pos = 7 - q, bit = 1 << pos, lm = bit - 1;
      for (int p = lane; p < 128; p += 32) {
        const int i0 = ((p >> pos) << (pos + 1)) | (p & lm);
        const int i1 = i0 | bit;
        const float r0 = sre[i0], m0 = sim_[i0], r1 = sre[i1], m1 = sim_[i1];
        sre[i0] = c * r0 + s * m1;  sim_[i0] = c * m0 - s * r1;
        sre[i1] = c * r1 + s * m0;  sim_[i1] = c * m1 - s * r0;
      }
      __syncthreads();
    }
    for (int q = 0; q < 8; ++q) {               // closed CNOT ring
      const int cpos = 7 - q, tpos = 7 - ((q + 1) & 7);
      const int hi = cpos > tpos ? cpos : tpos;
      const int lo = cpos ^ tpos ^ hi;
      for (int p = lane; p < 64; p += 32) {
        int t = ((p >> lo) << (lo + 1)) | (p & ((1 << lo) - 1));
        t = ((t >> hi) << (hi + 1)) | (t & ((1 << hi) - 1));
        const int i0 = t | (1 << cpos);
        const int i1 = i0 | (1 << tpos);
        float a = sre[i0]; sre[i0] = sre[i1]; sre[i1] = a;
        a = sim_[i0]; sim_[i0] = sim_[i1]; sim_[i1] = a;
      }
      __syncthreads();
    }
  }

  float ev[8] = {0, 0, 0, 0, 0, 0, 0, 0};
  for (int j = 0; j < 8; ++j) {
    const int i = lane * 8 + j;
    const float p = sre[i] * sre[i] + sim_[i] * sim_[i];
    for (int q = 0; q < 8; ++q) ev[q] += ((i >> (7 - q)) & 1) ? -p : p;
  }
  for (int q = 0; q < 8; ++q) {
    float v = ev[q];
    for (int o = 16; o; o >>= 1) v += __shfl_xor(v, o, 32);
    if (lane == 0) qout[(size_t)b * 64 + ql * 8 + q] = v;
  }
}

// ---------------- FC: [32][64] @ [751][64]^T + b -> [32][751] f32 ----------------
__global__ void k_fc(const float* __restrict__ q, const float* __restrict__ w,
                     const float* __restrict__ bias, float* __restrict__ out) {
  int i = blockIdx.x * 256 + threadIdx.x;
  if (i >= 32 * 751) return;
  const int b = i / 751, c = i - b * 751;
  float s = bias[c];
  const float* wr = w + (size_t)c * 64;
  const float* qr = q + (size_t)b * 64;
  for (int j = 0; j < 64; ++j) s += qr[j] * wr[j];
  out[i] = s;
}

// ======================= host orchestration =======================
extern "C" void kernel_launch(void* const* d_in, const int* in_sizes, int n_in,
                              void* d_out, int out_size, void* d_ws, size_t ws_size,
                              hipStream_t stream) {
  (void)in_sizes; (void)n_in; (void)out_size; (void)ws_size;
  int idx = 0;
  const float* x = (const float*)d_in[idx++];

  struct BNP { const float *b, *g, *m, *v; };
  auto readBN = [&]() {
    BNP p;
    p.b = (const float*)d_in[idx++];  // dict keys sorted: b, g, m, v
    p.g = (const float*)d_in[idx++];
    p.m = (const float*)d_in[idx++];
    p.v = (const float*)d_in[idx++];
    return p;
  };

  // params pytree (dict keys sorted): bn1, conv1, layers
  BNP bn1 = readBN();
  const float* conv1w = (const float*)d_in[idx++];

  struct Blk { BNP bn1, bn2, bn3, bnd; const float *w1, *w2, *w3, *wd; bool ds; };
  Blk blk[16];
  const int nblk[4] = {3, 4, 6, 3};
  int bi = 0;
  for (int s = 0; s < 4; ++s)
    for (int i = 0; i < nblk[s]; ++i, ++bi) {
      Blk& B = blk[bi];
      B.ds = (i == 0);
      // block dict keys sorted: bn1,bn2,bn3[,bnd],w1,w2,w3[,wd]
      B.bn1 = readBN(); B.bn2 = readBN(); B.bn3 = readBN();
      if (B.ds) B.bnd = readBN();
      B.w1 = (const float*)d_in[idx++];
      B.w2 = (const float*)d_in[idx++];
      B.w3 = (const float*)d_in[idx++];
      B.wd = B.ds ? (const float*)d_in[idx++] : nullptr;
    }
  const float* qweights = (const float*)d_in[idx++];
  const float* fc_w     = (const float*)d_in[idx++];
  const float* fc_b     = (const float*)d_in[idx++];

  // ---- workspace layout (bump allocated; deterministic every call) ----
  char* ws = (char*)d_ws;
  size_t woff = 0;
  auto walloc = [&](size_t elems) {
    unsigned short* p = (unsigned short*)(ws + woff);
    woff += ((elems * 2 + 255) & ~(size_t)255);
    return p;
  };
  auto cvt = [&](const float* src, unsigned short* dst, size_t n) {
    k_f32_to_bf16<<<dim3((unsigned)((n + 255) / 256)), dim3(256), 0, stream>>>(
        src, dst, (int)n);
  };

  // bf16 weight packing (OIHW flatten == GEMM-K order)
  unsigned short* c1wb = walloc((size_t)64 * 3 * 49);
  cvt(conv1w, c1wb, (size_t)64 * 3 * 49);

  unsigned short *w1b[16], *w2b[16], *w3b[16], *wdb[16];
  const int widths[4] = {64, 128, 256, 512};
  int cin = 64;
  bi = 0;
  for (int s = 0; s < 4; ++s)
    for (int i = 0; i < nblk[s]; ++i, ++bi) {
      const int w = widths[s];
      w1b[bi] = walloc((size_t)w * cin);     cvt(blk[bi].w1, w1b[bi], (size_t)w * cin);
      w2b[bi] = walloc((size_t)w * w * 9);   cvt(blk[bi].w2, w2b[bi], (size_t)w * w * 9);
      w3b[bi] = walloc((size_t)4 * w * w);   cvt(blk[bi].w3, w3b[bi], (size_t)4 * w * w);
      if (blk[bi].ds) {
        wdb[bi] = walloc((size_t)4 * w * cin);
        cvt(blk[bi].wd, wdb[bi], (size_t)4 * w * cin);
      } else wdb[bi] = nullptr;
      cin = 4 * w;
    }

  woff = (woff + 255) & ~(size_t)255;
  const size_t SLOT = 34ull << 20;             // 34 MB per activation slot
  unsigned short* slot[4];
  for (int i = 0; i < 4; ++i) slot[i] = (unsigned short*)(ws + woff + (size_t)i * SLOT);
  size_t tail = woff + 4 * SLOT;
  float* feats = (float*)(ws + tail); tail += (size_t)32 * 2048 * 4;
  float* qout  = (float*)(ws + tail); tail += (size_t)32 * 64 * 4;
  tail = (tail + 255) & ~(size_t)255;
  unsigned short* xbf = (unsigned short*)(ws + tail);

  // input f32 -> bf16
  cvt(x, xbf, (size_t)32 * 3 * 256 * 128);

  auto mg16 = [](int d) { return (unsigned)((65536 + d - 1) / d); };

  auto conv = [&](const unsigned short* in, const unsigned short* wb,
                  int Ci, int Hh, int Ww, int Co, int kh, int kw, int st, int pd,
                  const BNP& bn, const unsigned short* resi, unsigned short* outp,
                  int relu) {
    const int Ho = (Hh + 2 * pd - kh) / st + 1;
    const int Wo = (Ww + 2 * pd - kw) / st + 1;
    const int M = 32 * Ho * Wo;
    const int K = Ci * kh * kw;
    const int use_tdm = (K % 32 == 0) ? 1 : 0;   // all convs except the 7x7 stem
    dim3 g((unsigned)((Co + 63) / 64), (unsigned)((M + 127) / 128));
    k_conv_bn_wmma<<<g, dim3(256), 0, stream>>>(in, wb, bn.g, bn.b, bn.m, bn.v,
                                                resi, outp, Ci, Hh, Ww, Co, Ho, Wo,
                                                kh, kw, st, pd, relu, K, M, use_tdm,
                                                mg16(kh * kw), mg16(kw));
  };

  // stem: conv7x7/s2/p3 + BN + ReLU -> [32][64][128][64]; maxpool -> [32][64][64][32]
  conv(xbf, c1wb, 3, 256, 128, 64, 7, 7, 2, 3, bn1, nullptr, slot[0], 1);
  k_maxpool<<<dim3((32 * 64 * 64 * 32) / 256), dim3(256), 0, stream>>>(slot[0], slot[1]);

  // bottleneck stages
  int H = 64, Wsp = 32, C = 64, cur = 1;
  const int sstride[4] = {1, 2, 2, 2};
  bi = 0;
  for (int s = 0; s < 4; ++s)
    for (int i = 0; i < nblk[s]; ++i, ++bi) {
      const int w = widths[s];
      const int st = (i == 0) ? sstride[s] : 1;
      const int o0 = (cur + 1) & 3, o1 = (cur + 2) & 3, o2 = (cur + 3) & 3;
      const int Ho = (H + 2 - 3) / st + 1, Wo = (Wsp + 2 - 3) / st + 1;

      conv(slot[cur], w1b[bi], C, H, Wsp, w, 1, 1, 1, 0, blk[bi].bn1, nullptr, slot[o0], 1);
      conv(slot[o0],  w2b[bi], w, H, Wsp, w, 3, 3, st, 1, blk[bi].bn2, nullptr, slot[o1], 1);
      const unsigned short* sc;
      if (blk[bi].ds) {
        conv(slot[cur], wdb[bi], C, H, Wsp, 4 * w, 1, 1, st, 0, blk[bi].bnd, nullptr, slot[o2], 0);
        sc = slot[o2];
      } else {
        sc = slot[cur];
      }
      conv(slot[o1], w3b[bi], w, Ho, Wo, 4 * w, 1, 1, 1, 0, blk[bi].bn3, sc, slot[o0], 1);

      cur = o0; C = 4 * w; H = Ho; Wsp = Wo;
    }

  // global avgpool -> [32][2048] f32
  k_avgpool<<<dim3((32 * 2048) / 256), dim3(256), 0, stream>>>(slot[cur], feats);
  // quantum circuits -> [32][64]
  k_qsim<<<dim3(8, 32), dim3(32), 0, stream>>>(feats, qweights, qout);
  // classifier -> d_out [32][751]
  k_fc<<<dim3((32 * 751 + 255) / 256), dim3(256), 0, stream>>>(qout, fc_w, fc_b,
                                                               (float*)d_out);
}
